// MoE_44186623542092
// MI455X (gfx1250) — matrix operand order repaired and checked by reading, not compiled, not run on hardware
//
#include <hip/hip_runtime.h>
#include <hip/hip_bf16.h>
#include <math.h>

// ---------------------------------------------------------------------------
// MoE (top-2 routed SwiGLU experts + shared expert) for gfx1250 / MI455X.
// bf16 WMMA (v_wmma_f32_16x16x32_bf16) with fp32 accumulation.
// LDS tiles double-buffered; fragments are 2x ds_load_b128 each.
// FFN2 A-tiles are fetched with the Tensor Data Mover, overlapped with WMMA.
// ---------------------------------------------------------------------------

typedef __attribute__((ext_vector_type(16))) __bf16        v16bf;
typedef __attribute__((ext_vector_type(8)))  float         v8f;
typedef __attribute__((ext_vector_type(4)))  unsigned int  u32x4;
typedef __attribute__((ext_vector_type(8)))  int           i32x8;
typedef __attribute__((ext_vector_type(4)))  int           i32x4;

#define Bb  2
#define Tt  1024
#define Dd  1024
#define Ee  16
#define Hh  2048
#define Kk  2
#define HSs 4096
#define Nn  (Bb * Tt)      // 2048 tokens
#define NK  (Nn * Kk)      // 4096 assignments
#define CAP 5120           // NK + Ee*63 rounded up to 64

#if defined(__has_builtin)
#if __has_builtin(__builtin_amdgcn_tensor_load_to_lds)
#define HAVE_TDM 1
#endif
#endif
#ifndef HAVE_TDM
#define HAVE_TDM 0
#endif

__device__ __forceinline__ v8f wmma_bf16(v16bf a, v16bf b, v8f c) {
  return __builtin_amdgcn_wmma_f32_16x16x32_bf16(false, a, false, b, (short)0, c,
                                                 false, false);
}

// A fragment (16x32 bf16) from a row-major LDS tile [rows][32].
__device__ __forceinline__ v16bf load_a_frag_rm(const __bf16* lds, int row_base) {
  int lane = threadIdx.x & 31;
  int m = lane & 15, g = lane >> 4;
  const u32x4* p = (const u32x4*)(lds + (row_base + m) * 32);
  union { v16bf v; u32x4 q[2]; } t;
  t.q[0] = p[g];
  t.q[1] = p[2 + g];
  return t.v;
}

// B fragment (32x16 bf16) from a column-major LDS tile [64 cols][32 k].
__device__ __forceinline__ v16bf load_b_frag_cm(const __bf16* ldsT, int col_base) {
  int lane = threadIdx.x & 31;
  int n = lane & 15, kg = lane >> 4;
  const u32x4* p = (const u32x4*)(ldsT + (col_base + n) * 32);
  union { v16bf v; u32x4 q[2]; } t;
  t.q[0] = p[kg];
  t.q[1] = p[2 + kg];
  return t.v;
}

__device__ __forceinline__ unsigned pack_bf16x2(float f0, float f1) {
  union { __bf16 b; unsigned short u; } a, b;
  a.b = (__bf16)f0; b.b = (__bf16)f1;
  return (unsigned)a.u | ((unsigned)b.u << 16);
}

// Convert+transpose a 32x64 fp32 weight slab into LDS col-major [64][32] bf16.
__device__ __forceinline__ void load_b_tile_cm(__bf16* sBT, const float* wbase,
                                               int ld) {
  int c = threadIdx.x & 63;
  int kk = (threadIdx.x >> 6) * 2;
  const float* wp = wbase + (size_t)kk * ld + c;
#pragma unroll
  for (int i = 0; i < 8; ++i) {
    int k = i * 4 + kk;
    float f0 = wp[(size_t)(i * 4) * ld];
    float f1 = wp[(size_t)(i * 4 + 1) * ld];
    *(unsigned*)(sBT + c * 32 + k) = pack_bf16x2(f0, f1);
  }
}

__device__ __forceinline__ void load_b_tile_cm2(__bf16* sBT1, __bf16* sBT2,
                                                const float* w1base,
                                                const float* w2base, int ld) {
  int c = threadIdx.x & 63;
  int kk = (threadIdx.x >> 6) * 2;
  const float* p1 = w1base + (size_t)kk * ld + c;
  const float* p2 = w2base + (size_t)kk * ld + c;
#pragma unroll
  for (int i = 0; i < 8; ++i) {
    int k = i * 4 + kk;
    *(unsigned*)(sBT1 + c * 32 + k) =
        pack_bf16x2(p1[(size_t)(i * 4) * ld], p1[(size_t)(i * 4 + 1) * ld]);
    *(unsigned*)(sBT2 + c * 32 + k) =
        pack_bf16x2(p2[(size_t)(i * 4) * ld], p2[(size_t)(i * 4 + 1) * ld]);
  }
}

#if HAVE_TDM
// Tensor Data Mover: [tile_d1 rows x tile_d0 elems] bf16 tile, row stride
// stride_elems, into LDS byte offset lds_off. D# per CDNA5 ISA.
__device__ __forceinline__ void tdm_load_tile_bf16(unsigned lds_off, const void* gptr,
                                                   unsigned tensor_d0, unsigned tensor_d1,
                                                   unsigned tile_d0, unsigned tile_d1,
                                                   unsigned stride_elems) {
  unsigned long long ga = (unsigned long long)(size_t)gptr;
  u32x4 g0;
  g0[0] = 1u;                                         // count=1 (user descriptor)
  g0[1] = lds_off;                                    // lds_addr (bytes)
  g0[2] = (unsigned)(ga & 0xFFFFFFFFu);               // global_addr[31:0]
  g0[3] = (unsigned)((ga >> 32) & 0x01FFFFFFu) | (2u << 30);  // addr[56:32], type=2
  i32x8 g1;
  g1[0] = (int)(1u << 16);                            // workgroup_mask=0, data_size=2B
  g1[1] = (int)((tensor_d0 & 0xFFFFu) << 16);
  g1[2] = (int)((tensor_d0 >> 16) | ((tensor_d1 & 0xFFFFu) << 16));
  g1[3] = (int)((tensor_d1 >> 16) | (tile_d0 << 16));
  g1[4] = (int)(tile_d1 & 0xFFFFu);
  g1[5] = (int)stride_elems;                          // tensor_dim0_stride[31:0]
  g1[6] = 0;
  g1[7] = 0;
  i32x4 z4 = {0, 0, 0, 0};
#if __clang_major__ >= 23
  i32x8 z8 = {0, 0, 0, 0, 0, 0, 0, 0};
  __builtin_amdgcn_tensor_load_to_lds(g0, g1, z4, z4, z8, 0);
#else
  __builtin_amdgcn_tensor_load_to_lds(g0, g1, z4, z4, 0);
#endif
}
#endif

__device__ __forceinline__ float silu_f(float v) { return v / (1.f + __expf(-v)); }

// ---------------------------------------------------------------------------
// 1) init
// ---------------------------------------------------------------------------
__global__ void init_convert_kernel(const float* __restrict__ x,
                                    float* __restrict__ out,
                                    __bf16* __restrict__ xb,
                                    int* __restrict__ a_token,
                                    int* __restrict__ counts) {
  int idx = blockIdx.x * blockDim.x + threadIdx.x;
  if (idx < Nn * Dd) {
    out[idx] = 0.f;
    xb[idx] = (__bf16)x[idx];
  }
  if (idx < CAP) a_token[idx] = -1;
  if (idx < Ee)  counts[idx] = 0;
}

// ---------------------------------------------------------------------------
// 2) router
// ---------------------------------------------------------------------------
__global__ __launch_bounds__(128)
void router_topk_kernel(const float* __restrict__ x, const float* __restrict__ Wg,
                        int* __restrict__ topk_e, float* __restrict__ topk_g,
                        int* __restrict__ counts) {
  int lane = threadIdx.x & 31;
  int wave = threadIdx.x >> 5;
  int t = blockIdx.x * (blockDim.x >> 5) + wave;
  if (t >= Nn) return;

  float acc[Ee];
#pragma unroll
  for (int e = 0; e < Ee; ++e) acc[e] = 0.f;

  for (int d = lane; d < Dd; d += 32) {
    float xv = x[(size_t)t * Dd + d];
    const float* wr = Wg + (size_t)d * Ee;
#pragma unroll
    for (int e = 0; e < Ee; ++e) acc[e] += xv * wr[e];
  }
#pragma unroll
  for (int e = 0; e < Ee; ++e) {
#pragma unroll
    for (int m = 16; m >= 1; m >>= 1) acc[e] += __shfl_xor(acc[e], m, 32);
  }
  if (lane == 0) {
    int e0 = 0; float l0 = acc[0];
#pragma unroll
    for (int e = 1; e < Ee; ++e) if (acc[e] > l0) { l0 = acc[e]; e0 = e; }
    int e1 = (e0 == 0) ? 1 : 0; float l1 = acc[e1];
#pragma unroll
    for (int e = 0; e < Ee; ++e)
      if (e != e0 && acc[e] > l1) { l1 = acc[e]; e1 = e; }
    float g0 = 1.f / (1.f + __expf(l1 - l0));
    float g1 = 1.f - g0;
    topk_e[t * Kk + 0] = e0;  topk_g[t * Kk + 0] = g0;
    topk_e[t * Kk + 1] = e1;  topk_g[t * Kk + 1] = g1;
    atomicAdd(&counts[e0], 1);
    atomicAdd(&counts[e1], 1);
  }
}

// ---------------------------------------------------------------------------
// 3) scan
// ---------------------------------------------------------------------------
__global__ void scan_kernel(const int* __restrict__ counts,
                            int* __restrict__ p_off, int* __restrict__ cursor) {
  if (threadIdx.x == 0) {
    int acc = 0;
    for (int e = 0; e < Ee; ++e) {
      p_off[e] = acc;
      cursor[e] = acc;
      acc += (counts[e] + 63) & ~63;
    }
    p_off[Ee] = acc;
  }
}

// ---------------------------------------------------------------------------
// 4) compact assignments
// ---------------------------------------------------------------------------
__global__ void router_assign_kernel(const int* __restrict__ topk_e,
                                     const float* __restrict__ topk_g,
                                     int* __restrict__ cursor,
                                     int* __restrict__ a_token,
                                     float* __restrict__ a_gate) {
  int t = blockIdx.x * blockDim.x + threadIdx.x;
  if (t >= Nn) return;
#pragma unroll
  for (int j = 0; j < Kk; ++j) {
    int e = topk_e[t * Kk + j];
    int slot = atomicAdd(&cursor[e], 1);
    a_token[slot] = t;
    a_gate[slot] = topk_g[t * Kk + j];
  }
}

// ---------------------------------------------------------------------------
// 5) routed FFN1 (double-buffered)
// ---------------------------------------------------------------------------
__global__ __launch_bounds__(128)
void ffn1_routed_kernel(const __bf16* __restrict__ xb,
                        const float* __restrict__ W1, const float* __restrict__ b1,
                        const float* __restrict__ W11, const float* __restrict__ b11,
                        const int* __restrict__ p_off,
                        const int* __restrict__ a_token,
                        __bf16* __restrict__ h_routed) {
  __shared__ __bf16 sA[2][64 * 32];
  __shared__ __bf16 sB1T[2][64 * 32];
  __shared__ __bf16 sB11T[2][64 * 32];
  __shared__ int sTok[64];

  int row0 = blockIdx.x * 64;
  if (row0 >= p_off[Ee]) return;
  int e = 0;
#pragma unroll
  for (int i = 0; i < Ee; ++i) if (row0 >= p_off[i + 1]) e = i + 1;
  int h0 = blockIdx.y * 64;
  int tid = threadIdx.x;
  if (tid < 64) sTok[tid] = a_token[row0 + tid];
  __syncthreads();

  // loop-invariant gather state (2 chunks per thread)
  int r0 = tid >> 2, r1 = (tid + 128) >> 2;
  int c8 = (tid & 3) * 8;
  int tok0 = sTok[r0], tok1 = sTok[r1];
  const __bf16* src0 = xb + (size_t)(tok0 < 0 ? 0 : tok0) * Dd + c8;
  const __bf16* src1 = xb + (size_t)(tok1 < 0 ? 0 : tok1) * Dd + c8;
  const u32x4 zed = {0, 0, 0, 0};

  const float* w1  = W1  + (size_t)e * Dd * Hh + h0;
  const float* w11 = W11 + (size_t)e * Dd * Hh + h0;

  int wave = tid >> 5;
  v8f acc1[4], acc11[4];
#pragma unroll
  for (int j = 0; j < 4; ++j) { v8f z = {}; acc1[j] = z; acc11[j] = z; }

  // prologue: stage k0 = 0 into buffer 0
  *(u32x4*)(sA[0] + r0 * 32 + c8) = (tok0 >= 0) ? *(const u32x4*)(src0) : zed;
  *(u32x4*)(sA[0] + r1 * 32 + c8) = (tok1 >= 0) ? *(const u32x4*)(src1) : zed;
  load_b_tile_cm2(sB1T[0], sB11T[0], w1, w11, Hh);
  __syncthreads();

  const int nk = Dd / 32;
  for (int i = 0; i < nk; ++i) {
    int cur = i & 1, nxt = cur ^ 1;
    if (i + 1 < nk) {
      int k0 = (i + 1) * 32;
      *(u32x4*)(sA[nxt] + r0 * 32 + c8) = (tok0 >= 0) ? *(const u32x4*)(src0 + k0) : zed;
      *(u32x4*)(sA[nxt] + r1 * 32 + c8) = (tok1 >= 0) ? *(const u32x4*)(src1 + k0) : zed;
      load_b_tile_cm2(sB1T[nxt], sB11T[nxt], w1 + (size_t)k0 * Hh,
                      w11 + (size_t)k0 * Hh, Hh);
    }
    v16bf a = load_a_frag_rm(sA[cur], wave * 16);
#pragma unroll
    for (int j = 0; j < 4; ++j) {
      acc1[j]  = wmma_bf16(a, load_b_frag_cm(sB1T[cur],  j * 16), acc1[j]);
      acc11[j] = wmma_bf16(a, load_b_frag_cm(sB11T[cur], j * 16), acc11[j]);
    }
    __syncthreads();
  }

  int lane = tid & 31;
  int n_l = lane & 15;
  int m_hi = (lane >> 4) * 8;
#pragma unroll
  for (int j = 0; j < 4; ++j) {
#pragma unroll
    for (int v = 0; v < 8; ++v) {
      int row = row0 + wave * 16 + m_hi + v;
      int col = h0 + j * 16 + n_l;
      float g1 = acc1[j][v]  + b1[e * Hh + col];
      float g2 = acc11[j][v] + b11[e * Hh + col];
      h_routed[(size_t)row * Hh + col] = (__bf16)(silu_f(g1) * g2);
    }
  }
}

// ---------------------------------------------------------------------------
// 6) routed FFN2: A tiles via TDM, double-buffered against WMMA.
//    Dynamic LDS: sA0@0, sA1@4096, sBT0@8192, sBT1@12288, sTok@16384, sGate@16640
// ---------------------------------------------------------------------------
__global__ __launch_bounds__(128)
void ffn2_routed_kernel(const __bf16* __restrict__ h_routed,
                        const float* __restrict__ W2, const float* __restrict__ b2,
                        const int* __restrict__ p_off,
                        const int* __restrict__ a_token,
                        const float* __restrict__ a_gate,
                        float* __restrict__ out) {
  extern __shared__ char smem[];
  __bf16* sA0  = (__bf16*)smem;
  __bf16* sA1  = (__bf16*)(smem + 4096);
  __bf16* sBT0 = (__bf16*)(smem + 8192);
  __bf16* sBT1 = (__bf16*)(smem + 12288);
  int*    sTok  = (int*)(smem + 16384);
  float*  sGate = (float*)(smem + 16640);

  int row0 = blockIdx.x * 64;
  if (row0 >= p_off[Ee]) return;
  int e = 0;
#pragma unroll
  for (int i = 0; i < Ee; ++i) if (row0 >= p_off[i + 1]) e = i + 1;
  int d0 = blockIdx.y * 64;
  int tid = threadIdx.x;
  if (tid < 64) { sTok[tid] = a_token[row0 + tid]; sGate[tid] = a_gate[row0 + tid]; }

  const __bf16* abase = h_routed + (size_t)row0 * Hh;
  const float* w2 = W2 + (size_t)e * Hh * Dd + d0;
  int wave = tid >> 5;
  v8f acc[4];
#pragma unroll
  for (int j = 0; j < 4; ++j) { v8f z = {}; acc[j] = z; }

  // prologue: stage tile 0
#if HAVE_TDM
  if (wave == 0) {
    tdm_load_tile_bf16(0u, abase, Hh, 64, 32, 64, Hh);
    __builtin_amdgcn_s_wait_tensorcnt(0);
  }
#else
  {
    int r = tid >> 2, c8 = (tid & 3) * 8;
    *(u32x4*)(sA0 + r * 32 + c8) = *(const u32x4*)(abase + (size_t)r * Hh + c8);
    r = (tid + 128) >> 2;
    *(u32x4*)(sA0 + r * 32 + c8) = *(const u32x4*)(abase + (size_t)r * Hh + c8);
  }
#endif
  load_b_tile_cm(sBT0, w2, Dd);
  __syncthreads();

  const int nk = Hh / 32;
  for (int i = 0; i < nk; ++i) {
    int cur = i & 1;
    __bf16* sAc = cur ? sA1 : sA0;
    __bf16* sBc = cur ? sBT1 : sBT0;
    __bf16* sBn = cur ? sBT0 : sBT1;
    if (i + 1 < nk) {
      int k0 = (i + 1) * 32;
#if HAVE_TDM
      if (wave == 0)
        tdm_load_tile_bf16(cur ? 0u : 4096u, abase + k0, Hh, 64, 32, 64, Hh);
#else
      {
        __bf16* sAn = cur ? sA0 : sA1;
        int r = tid >> 2, c8 = (tid & 3) * 8;
        *(u32x4*)(sAn + r * 32 + c8) =
            *(const u32x4*)(abase + (size_t)r * Hh + k0 + c8);
        r = (tid + 128) >> 2;
        *(u32x4*)(sAn + r * 32 + c8) =
            *(const u32x4*)(abase + (size_t)r * Hh + k0 + c8);
      }
#endif
      load_b_tile_cm(sBn, w2 + (size_t)k0 * Dd, Dd);
    }
    v16bf a = load_a_frag_rm(sAc, wave * 16);
#pragma unroll
    for (int j = 0; j < 4; ++j)
      acc[j] = wmma_bf16(a, load_b_frag_cm(sBc, j * 16), acc[j]);
#if HAVE_TDM
    if (wave == 0 && i + 1 < nk) __builtin_amdgcn_s_wait_tensorcnt(0);
#endif
    __syncthreads();
  }

  int lane = tid & 31;
  int n_l = lane & 15;
  int m_hi = (lane >> 4) * 8;
#pragma unroll
  for (int j = 0; j < 4; ++j) {
#pragma unroll
    for (int v = 0; v < 8; ++v) {
      int rl = wave * 16 + m_hi + v;
      int tok = sTok[rl];
      if (tok >= 0) {
        int col = d0 + j * 16 + n_l;
        atomicAdd(&out[(size_t)tok * Dd + col],
                  sGate[rl] * (acc[j][v] + b2[e * Dd + col]));
      }
    }
  }
}

// ---------------------------------------------------------------------------
// 7) shared FFN1 (double-buffered)
// ---------------------------------------------------------------------------
__global__ __launch_bounds__(128)
void ffn1_shared_kernel(const __bf16* __restrict__ xb,
                        const float* __restrict__ Ws1, const float* __restrict__ bs1,
                        const float* __restrict__ Ws11, const float* __restrict__ bs11,
                        __bf16* __restrict__ h_shared) {
  __shared__ __bf16 sA[2][64 * 32];
  __shared__ __bf16 sB1T[2][64 * 32];
  __shared__ __bf16 sB11T[2][64 * 32];

  int row0 = blockIdx.x * 64;
  int h0 = blockIdx.y * 64;
  int tid = threadIdx.x;
  int wave = tid >> 5;

  int r0 = tid >> 2, r1 = (tid + 128) >> 2;
  int c8 = (tid & 3) * 8;
  const __bf16* src0 = xb + (size_t)(row0 + r0) * Dd + c8;
  const __bf16* src1 = xb + (size_t)(row0 + r1) * Dd + c8;
  const float* w1  = Ws1  + h0;
  const float* w11 = Ws11 + h0;

  v8f acc1[4], acc11[4];
#pragma unroll
  for (int j = 0; j < 4; ++j) { v8f z = {}; acc1[j] = z; acc11[j] = z; }

  *(u32x4*)(sA[0] + r0 * 32 + c8) = *(const u32x4*)(src0);
  *(u32x4*)(sA[0] + r1 * 32 + c8) = *(const u32x4*)(src1);
  load_b_tile_cm2(sB1T[0], sB11T[0], w1, w11, HSs);
  __syncthreads();

  const int nk = Dd / 32;
  for (int i = 0; i < nk; ++i) {
    int cur = i & 1, nxt = cur ^ 1;
    if (i + 1 < nk) {
      int k0 = (i + 1) * 32;
      *(u32x4*)(sA[nxt] + r0 * 32 + c8) = *(const u32x4*)(src0 + k0);
      *(u32x4*)(sA[nxt] + r1 * 32 + c8) = *(const u32x4*)(src1 + k0);
      load_b_tile_cm2(sB1T[nxt], sB11T[nxt], w1 + (size_t)k0 * HSs,
                      w11 + (size_t)k0 * HSs, HSs);
    }
    v16bf a = load_a_frag_rm(sA[cur], wave * 16);
#pragma unroll
    for (int j = 0; j < 4; ++j) {
      acc1[j]  = wmma_bf16(a, load_b_frag_cm(sB1T[cur],  j * 16), acc1[j]);
      acc11[j] = wmma_bf16(a, load_b_frag_cm(sB11T[cur], j * 16), acc11[j]);
    }
    __syncthreads();
  }

  int lane = tid & 31;
  int n_l = lane & 15;
  int m_hi = (lane >> 4) * 8;
#pragma unroll
  for (int j = 0; j < 4; ++j) {
#pragma unroll
    for (int v = 0; v < 8; ++v) {
      int row = row0 + wave * 16 + m_hi + v;
      int col = h0 + j * 16 + n_l;
      float g1 = acc1[j][v]  + bs1[col];
      float g2 = acc11[j][v] + bs11[col];
      h_shared[(size_t)row * HSs + col] = (__bf16)(silu_f(g1) * g2);
    }
  }
}

// ---------------------------------------------------------------------------
// 8) shared FFN2: A tiles via TDM, double-buffered.
//    Dynamic LDS: sA0@0, sA1@4096, sBT0@8192, sBT1@12288
// ---------------------------------------------------------------------------
__global__ __launch_bounds__(128)
void ffn2_shared_kernel(const __bf16* __restrict__ h_shared,
                        const float* __restrict__ Ws2, const float* __restrict__ bs2,
                        float* __restrict__ out) {
  extern __shared__ char smem[];
  __bf16* sA0  = (__bf16*)smem;
  __bf16* sA1  = (__bf16*)(smem + 4096);
  __bf16* sBT0 = (__bf16*)(smem + 8192);
  __bf16* sBT1 = (__bf16*)(smem + 12288);

  int row0 = blockIdx.x * 64;
  int d0 = blockIdx.y * 64;
  int tid = threadIdx.x;
  int wave = tid >> 5;
  const __bf16* abase = h_shared + (size_t)row0 * HSs;
  const float* w2 = Ws2 + d0;
  v8f acc[4];
#pragma unroll
  for (int j = 0; j < 4; ++j) { v8f z = {}; acc[j] = z; }

#if HAVE_TDM
  if (wave == 0) {
    tdm_load_tile_bf16(0u, abase, HSs, 64, 32, 64, HSs);
    __builtin_amdgcn_s_wait_tensorcnt(0);
  }
#else
  {
    int r = tid >> 2, c8 = (tid & 3) * 8;
    *(u32x4*)(sA0 + r * 32 + c8) = *(const u32x4*)(abase + (size_t)r * HSs + c8);
    r = (tid + 128) >> 2;
    *(u32x4*)(sA0 + r * 32 + c8) = *(const u32x4*)(abase + (size_t)r * HSs + c8);
  }
#endif
  load_b_tile_cm(sBT0, w2, Dd);
  __syncthreads();

  const int nk = HSs / 32;
  for (int i = 0; i < nk; ++i) {
    int cur = i & 1;
    __bf16* sAc = cur ? sA1 : sA0;
    __bf16* sBc = cur ? sBT1 : sBT0;
    __bf16* sBn = cur ? sBT0 : sBT1;
    if (i + 1 < nk) {
      int k0 = (i + 1) * 32;
#if HAVE_TDM
      if (wave == 0)
        tdm_load_tile_bf16(cur ? 0u : 4096u, abase + k0, HSs, 64, 32, 64, HSs);
#else
      {
        __bf16* sAn = cur ? sA0 : sA1;
        int r = tid >> 2, c8 = (tid & 3) * 8;
        *(u32x4*)(sAn + r * 32 + c8) =
            *(const u32x4*)(abase + (size_t)r * HSs + k0 + c8);
        r = (tid + 128) >> 2;
        *(u32x4*)(sAn + r * 32 + c8) =
            *(const u32x4*)(abase + (size_t)r * HSs + k0 + c8);
      }
#endif
      load_b_tile_cm(sBn, w2 + (size_t)k0 * Dd, Dd);
    }
    v16bf a = load_a_frag_rm(sAc, wave * 16);
#pragma unroll
    for (int j = 0; j < 4; ++j)
      acc[j] = wmma_bf16(a, load_b_frag_cm(sBc, j * 16), acc[j]);
#if HAVE_TDM
    if (wave == 0 && i + 1 < nk) __builtin_amdgcn_s_wait_tensorcnt(0);
#endif
    __syncthreads();
  }

  int lane = tid & 31;
  int n_l = lane & 15;
  int m_hi = (lane >> 4) * 8;
#pragma unroll
  for (int j = 0; j < 4; ++j) {
#pragma unroll
    for (int v = 0; v < 8; ++v) {
      int row = row0 + wave * 16 + m_hi + v;
      int col = d0 + j * 16 + n_l;
      out[(size_t)row * Dd + col] += acc[j][v] + bs2[col];
    }
  }
}

// ---------------------------------------------------------------------------
// launch
// ---------------------------------------------------------------------------
extern "C" void kernel_launch(void* const* d_in, const int* in_sizes, int n_in,
                              void* d_out, int out_size, void* d_ws, size_t ws_size,
                              hipStream_t stream) {
  (void)in_sizes; (void)n_in; (void)out_size; (void)ws_size;
  const float* x    = (const float*)d_in[0];
  const float* Wg   = (const float*)d_in[1];
  const float* W1   = (const float*)d_in[2];
  const float* b1   = (const float*)d_in[3];
  const float* W11  = (const float*)d_in[4];
  const float* b11  = (const float*)d_in[5];
  const float* W2   = (const float*)d_in[6];
  const float* b2   = (const float*)d_in[7];
  const float* Ws1  = (const float*)d_in[8];
  const float* bs1  = (const float*)d_in[9];
  const float* Ws11 = (const float*)d_in[10];
  const float* bs11 = (const float*)d_in[11];
  const float* Ws2  = (const float*)d_in[12];
  const float* bs2  = (const float*)d_in[13];
  float* out = (float*)d_out;

  char* ws = (char*)d_ws;
  size_t off = 0;
  auto alignup = [](size_t v) { return (v + 255) & ~(size_t)255; };
  int*    counts   = (int*)(ws + off);    off = alignup(off + Ee * 4);
  int*    p_off    = (int*)(ws + off);    off = alignup(off + (Ee + 1) * 4);
  int*    cursor   = (int*)(ws + off);    off = alignup(off + Ee * 4);
  int*    topk_e   = (int*)(ws + off);    off = alignup(off + NK * 4);
  float*  topk_g   = (float*)(ws + off);  off = alignup(off + NK * 4);
  int*    a_token  = (int*)(ws + off);    off = alignup(off + CAP * 4);
  float*  a_gate   = (float*)(ws + off);  off = alignup(off + CAP * 4);
  __bf16* xb       = (__bf16*)(ws + off); off = alignup(off + (size_t)Nn * Dd * 2);
  __bf16* h_routed = (__bf16*)(ws + off); off = alignup(off + (size_t)CAP * Hh * 2);
  __bf16* h_shared = (__bf16*)(ws + off); off = alignup(off + (size_t)Nn * HSs * 2);

  init_convert_kernel<<<(Nn * Dd + 255) / 256, 256, 0, stream>>>(x, out, xb, a_token, counts);
  router_topk_kernel<<<Nn / 4, 128, 0, stream>>>(x, Wg, topk_e, topk_g, counts);
  scan_kernel<<<1, 32, 0, stream>>>(counts, p_off, cursor);
  router_assign_kernel<<<(Nn + 255) / 256, 256, 0, stream>>>(topk_e, topk_g, cursor,
                                                             a_token, a_gate);
  ffn1_routed_kernel<<<dim3(CAP / 64, Hh / 64), 128, 0, stream>>>(
      xb, W1, b1, W11, b11, p_off, a_token, h_routed);
  ffn2_routed_kernel<<<dim3(CAP / 64, Dd / 64), 128, 16896, stream>>>(
      h_routed, W2, b2, p_off, a_token, a_gate, out);
  ffn1_shared_kernel<<<dim3(Nn / 64, HSs / 64), 128, 0, stream>>>(
      xb, Ws1, bs1, Ws11, bs11, h_shared);
  ffn2_shared_kernel<<<dim3(Nn / 64, Dd / 64), 128, 16384, stream>>>(
      h_shared, Ws2, bs2, out);
}
